// RNN_15247133901472
// MI455X (gfx1250) — compile-verified, matrix operand order checked by reading, not run on hardware
//
#include <hip/hip_runtime.h>
#include <hip/hip_bf16.h>

// Problem constants (reference): T=1024, B=128, F=88, H=1024, C=88
#define TT 1024
#define BB 128
#define FF 88
#define HH 1024
#define CC 88
#define NWG 64            // persistent workgroups for the recurrence

typedef __attribute__((ext_vector_type(16))) __bf16 v16bf;
typedef __attribute__((ext_vector_type(8)))  float  v8f;
typedef __attribute__((ext_vector_type(4)))  unsigned u32x4;
typedef __attribute__((ext_vector_type(4)))  int      i32x4;
typedef __attribute__((ext_vector_type(8)))  int      i32x8;

// Fragment counts
#define WHH_FRAGS (4 * 32 * 32)   // 4096 v16bf per WG
#define WIH_FRAGS (4 * 3 * 32)    // 384  v16bf per WG
#define REC_FRAGS (WHH_FRAGS + WIH_FRAGS)       // 4480 -> 143,360 B per WG
#define FC_FRAGS  (6 * 32 * 32)                 // 6144 -> 196,608 B

// ---------------------------------------------------------------------------
// TDM: 1D block DMA global -> LDS via Tensor Data Mover (D# per ISA ch.8).
//   gaddr      : global byte address of block start
//   lds_off    : byte offset within this WG's LDS allocation
//   nelem8     : number of 8-byte elements (tile_dim0 is 16-bit: < 65536)
// Waits handled by caller via s_wait_tensorcnt.
// ---------------------------------------------------------------------------
#if __has_builtin(__builtin_amdgcn_tensor_load_to_lds)
#define HAVE_TDM 1
__device__ __forceinline__ void tdm_load_1d(const void* gaddr,
                                            unsigned lds_off,
                                            unsigned nelem8) {
  unsigned long long ga = (unsigned long long)gaddr;
  u32x4 g0;
  g0[0] = 1u;                                            // count=1, user D#
  g0[1] = lds_off;                                       // lds_addr
  g0[2] = (unsigned)(ga & 0xffffffffu);                  // global_addr[31:0]
  g0[3] = (unsigned)((ga >> 32) & 0x01ffffffu)           // global_addr[56:32]
          | (2u << 30);                                  // type = 2 (image)
  i32x8 g1;
  g1[0] = (int)(3u << 16);                               // data_size = 8 bytes
  g1[1] = (int)((nelem8 & 0xffffu) << 16);               // tensor_dim0[15:0]
  g1[2] = (int)(((nelem8 >> 16) & 0xffffu) | (1u << 16));// dim0[31:16], dim1=1
  g1[3] = (int)((nelem8 & 0xffffu) << 16);               // tile_dim0
  g1[4] = 1;                                             // tile_dim1=1
  g1[5] = (int)nelem8;                                   // dim0_stride[31:0]
  g1[6] = 0;
  g1[7] = 0;
  i32x4 g2 = {0, 0, 0, 0};                               // dims 2/3 unused
  i32x4 g3 = {0, 0, 0, 0};
#if __has_include(<hip/amd_detail/amd_gfx1250_TDM.h>)
  i32x8 g4 = {0, 0, 0, 0, 0, 0, 0, 0};
  __builtin_amdgcn_tensor_load_to_lds(g0, g1, g2, g3, g4, 0);   // 6-arg lane
#else
  __builtin_amdgcn_tensor_load_to_lds(g0, g1, g2, g3, 0);       // 5-arg lane
#endif
}
__device__ __forceinline__ void tdm_wait0() {
#if __has_builtin(__builtin_amdgcn_s_wait_tensorcnt)
  __builtin_amdgcn_s_wait_tensorcnt(0);
#else
  asm volatile("s_wait_tensorcnt 0x0" ::: "memory");
#endif
}
#else
#define HAVE_TDM 0
#endif

// ---------------------------------------------------------------------------
// Prep A: zero h_0 (B x H bf16) and the grid-barrier counter.
// ---------------------------------------------------------------------------
__global__ void lstm_prep(__bf16* __restrict__ hbuf, unsigned* __restrict__ ctr) {
  int idx = blockIdx.x * blockDim.x + threadIdx.x;
  if (idx == 0) *ctr = 0u;
  if (idx < BB * HH) hbuf[idx] = (__bf16)0.0f;
}

// ---------------------------------------------------------------------------
// Prep B: pack W_hh / W_ih (f32) into per-WG bf16 WMMA B-fragment blocks.
//   Block cc (contiguous 143,360 B): [4 g][32 kc][32 lanes] for W_hh slice,
//   then [4 g][3 kc][32 lanes] for W_ih slice (K padded 88 -> 96).
//   B 32x16 fragment: lane l holds col n=l&15, K-half (l>>4)*16, 16 contig K.
// ---------------------------------------------------------------------------
__global__ void pack_rec_weights(const float* __restrict__ Whh,  // [4H][H]
                                 const float* __restrict__ Wih,  // [4H][F]
                                 v16bf* __restrict__ wpack)      // [64][4480]
{
  int e = blockIdx.x * blockDim.x + threadIdx.x;
  if (e >= NWG * REC_FRAGS) return;
  int cc = e / REC_FRAGS;
  int r  = e % REC_FRAGS;
  v16bf v;
  if (r < WHH_FRAGS) {
    int g  = r >> 10;
    int kc = (r >> 5) & 31;
    int l  = r & 31;
    int n  = l & 15;
    int kb = kc * 32 + ((l >> 4) << 4);
    const float* src = Whh + (size_t)(g * HH + cc * 16 + n) * HH + kb;
#pragma unroll
    for (int j = 0; j < 16; ++j) v[j] = (__bf16)src[j];
  } else {
    int r2 = r - WHH_FRAGS;
    int g  = r2 / 96;
    int kc = (r2 / 32) % 3;
    int l  = r2 & 31;
    int n  = l & 15;
    int kb = kc * 32 + ((l >> 4) << 4);
    const float* src = Wih + (size_t)(g * HH + cc * 16 + n) * FF;
#pragma unroll
    for (int j = 0; j < 16; ++j) {
      int k = kb + j;
      v[j] = (__bf16)((k < FF) ? src[k] : 0.0f);
    }
  }
  wpack[e] = v;
}

// ---------------------------------------------------------------------------
// Prep C: pack W_fc (f32 [C][H]) into bf16 B-fragments [6 ct][32 kc][32 l],
// cols padded 88 -> 96 with zeros.
// ---------------------------------------------------------------------------
__global__ void pack_fc_weights(const float* __restrict__ Wfc,
                                v16bf* __restrict__ fpack) {
  int e = blockIdx.x * blockDim.x + threadIdx.x;
  if (e >= FC_FRAGS) return;
  int ct = e >> 10;
  int kc = (e >> 5) & 31;
  int l  = e & 31;
  int nn = ct * 16 + (l & 15);
  int kb = kc * 32 + ((l >> 4) << 4);
  v16bf v;
#pragma unroll
  for (int j = 0; j < 16; ++j)
    v[j] = (__bf16)((nn < CC) ? Wfc[(size_t)nn * HH + kb + j] : 0.0f);
  fpack[e] = v;
}

// ---------------------------------------------------------------------------
// Stage a packed fragment block into LDS: TDM DMA if available, else
// cooperative vector copy. Ends with all threads synchronized on valid LDS.
// ---------------------------------------------------------------------------
__device__ __forceinline__ void stage_lds(v16bf* __restrict__ dst,
                                          const v16bf* __restrict__ src,
                                          int nfrag, int tid, int wid) {
#if HAVE_TDM
  if (wid == 0) {
    tdm_load_1d((const void*)src, 0u, (unsigned)(nfrag * 4)); // 32B = 4 x 8B
    tdm_wait0();
  }
  (void)dst; (void)tid;
#else
  for (int i = tid; i < nfrag; i += 256) dst[i] = src[i];
#endif
  __syncthreads();
}

// ---------------------------------------------------------------------------
// Persistent LSTM recurrence.
//   grid = 64 WGs x 256 threads (8 waves). WG `cc` owns h-columns
//   [cc*16, cc*16+16) => gate columns {g*1024 + cc*16 .. +16}, g in {i,f,g,o}.
//   W slices live in LDS (TDM-loaded once) for all 1024 steps. Wave r owns
//   batch row-tile r; its c/h cell state stays in VGPRs for the whole kernel.
//   hbuf[t] = h before step t (bf16 [T+1][B][H]); hbuf[0] zeroed by prep.
// ---------------------------------------------------------------------------
__global__ void lstm_persistent(const float* __restrict__ x,      // [T][B][F]
                                const int*   __restrict__ lens,   // [B]
                                const float* __restrict__ bih,    // [4H]
                                const float* __restrict__ bhh,    // [4H]
                                const v16bf* __restrict__ wpack,  // [64][4480]
                                __bf16*      __restrict__ hbuf,   // [T+1][B][H]
                                unsigned*    __restrict__ ctr)
{
  extern __shared__ __align__(32) unsigned char smem_raw[];
  v16bf* sWhh = (v16bf*)smem_raw;         // [4][32 kc][32 lanes]
  v16bf* sWih = sWhh + WHH_FRAGS;         // [4][3  kc][32 lanes]

  const int tid  = threadIdx.x;
  const int lane = tid & 31;
  const int wid  = tid >> 5;              // wave id == batch row-tile 0..7
  const int cc   = blockIdx.x;            // column owner 0..63

  stage_lds(sWhh, wpack + (size_t)cc * REC_FRAGS, REC_FRAGS, tid, wid);

  // Per-lane constants.
  const int n    = lane & 15;
  const int colh = cc * 16 + n;                   // owned h column (D: N=lane&15)
  float bias[4];
#pragma unroll
  for (int g = 0; g < 4; ++g) bias[g] = bih[g * HH + colh] + bhh[g * HH + colh];

  const int mofs = wid * 16 + ((lane >> 4) << 3); // D rows: m = mofs + v
  int lenm[8];
#pragma unroll
  for (int v = 0; v < 8; ++v) lenm[v] = lens[mofs + v];

  const int mA    = wid * 16 + n;                 // A rows: lane = M
  const int kofsA = (lane >> 4) << 4;             // A K-half per lane

  float hreg[8], creg[8];
#pragma unroll
  for (int v = 0; v < 8; ++v) { hreg[v] = 0.0f; creg[v] = 0.0f; }

  for (int t = 0; t < TT; ++t) {
    v8f acc[4];
#pragma unroll
    for (int g = 0; g < 4; ++g)
#pragma unroll
      for (int v = 0; v < 8; ++v) acc[g][v] = bias[g];

    // Warm next step's x rows into cache while we compute.
    if (t + 1 < TT)
      __builtin_prefetch(x + ((size_t)(t + 1) * BB + mA) * FF, 0, 1);

    // gates += h_t @ W_hh^T  (K = 1024 -> 32 bf16 WMMA steps, B from LDS)
    const __bf16* hrow = hbuf + ((size_t)t * BB + mA) * HH + kofsA;
#pragma unroll 4
    for (int kc = 0; kc < 32; ++kc) {
      v16bf a = *(const v16bf*)(hrow + kc * 32);
#pragma unroll
      for (int g = 0; g < 4; ++g)
        acc[g] = __builtin_amdgcn_wmma_f32_16x16x32_bf16(
            false, a, false, sWhh[(g * 32 + kc) * 32 + lane],
            (short)0, acc[g], false, false);
    }

    // gates += x_t @ W_ih^T  (K = 88 padded to 96 -> 3 WMMA steps)
    const float* xrow = x + ((size_t)t * BB + mA) * FF;
#pragma unroll
    for (int kc = 0; kc < 3; ++kc) {
      v16bf a;
#pragma unroll
      for (int j = 0; j < 16; ++j) {
        int k = kc * 32 + kofsA + j;
        a[j] = (__bf16)((k < FF) ? xrow[k] : 0.0f);
      }
#pragma unroll
      for (int g = 0; g < 4; ++g)
        acc[g] = __builtin_amdgcn_wmma_f32_16x16x32_bf16(
            false, a, false, sWih[(g * 3 + kc) * 32 + lane],
            (short)0, acc[g], false, false);
    }

    // LSTM cell (fp32 state in VGPRs), length mask, publish h_{t+1} as bf16.
    __bf16* hout = hbuf + (size_t)(t + 1) * BB * HH;
#pragma unroll
    for (int v = 0; v < 8; ++v) {
      float gi = 1.0f / (1.0f + __expf(-acc[0][v]));
      float gf = 1.0f / (1.0f + __expf(-acc[1][v]));
      float gg = tanhf(acc[2][v]);
      float go = 1.0f / (1.0f + __expf(-acc[3][v]));
      float cn = gf * creg[v] + gi * gg;
      float hn = go * tanhf(cn);
      if (t < lenm[v]) { creg[v] = cn; hreg[v] = hn; }
      hout[(size_t)(mofs + v) * HH + colh] = (__bf16)hreg[v];
    }

    // Grid-wide barrier: all 64 WGs must publish h_{t+1} before step t+1.
    __threadfence();
    __syncthreads();
    if (tid == 0) {
      atomicAdd(ctr, 1u);
      const unsigned target = (unsigned)NWG * (unsigned)(t + 1);
      while (__hip_atomic_load(ctr, __ATOMIC_ACQUIRE, __HIP_MEMORY_SCOPE_AGENT) < target)
        __builtin_amdgcn_s_sleep(2);
    }
    __syncthreads();
  }
}

// ---------------------------------------------------------------------------
// FC head: logits = h_hist @ W_fc^T + b_fc, masked, expanded to (v, 1-v).
//   Big-matrix rows R = t*B + b (131072 rows), stored at hbuf row R + B.
//   grid = 1024 WGs x 256; each wave: one 16-row tile x 6 col-tiles (96
//   padded cols, only n<88 written). W_fc fragments TDM-loaded into LDS.
// ---------------------------------------------------------------------------
__global__ void lstm_logits(const __bf16* __restrict__ hbuf,   // [T+1][B][H]
                            const int*    __restrict__ lens,
                            const v16bf*  __restrict__ fpack,  // [6144]
                            const float*  __restrict__ bfc,    // [C]
                            float*        __restrict__ out)    // [B*T*C][2]
{
  extern __shared__ __align__(32) unsigned char smem_raw[];
  v16bf* sW = (v16bf*)smem_raw;            // [6 ct][32 kc][32 lanes]

  const int tid = threadIdx.x, lane = tid & 31, wid = tid >> 5;

  stage_lds(sW, fpack, FC_FRAGS, tid, wid);

  const int RT    = blockIdx.x * 8 + wid;   // row-tile 0..8191
  const int n     = lane & 15;
  const int mA    = RT * 16 + n;
  const int kofsA = (lane >> 4) << 4;

  v8f acc[6];
#pragma unroll
  for (int ct = 0; ct < 6; ++ct) {
    int nn = ct * 16 + n;
    float bv = (nn < CC) ? bfc[nn] : 0.0f;
#pragma unroll
    for (int v = 0; v < 8; ++v) acc[ct][v] = bv;
  }

  const __bf16* hrow = hbuf + ((size_t)BB + mA) * HH + kofsA;  // skip h_0 block
#pragma unroll 4
  for (int kc = 0; kc < 32; ++kc) {
    v16bf a = *(const v16bf*)(hrow + kc * 32);
#pragma unroll
    for (int ct = 0; ct < 6; ++ct)
      acc[ct] = __builtin_amdgcn_wmma_f32_16x16x32_bf16(
          false, a, false, sW[(ct * 32 + kc) * 32 + lane],
          (short)0, acc[ct], false, false);
  }

  // Epilogue: mask by length, write (v, 1-v) pairs (contiguous float2).
#pragma unroll
  for (int v = 0; v < 8; ++v) {
    int R = RT * 16 + v + ((lane >> 4) << 3);
    int t = R >> 7;      // / B
    int b = R & 127;     // % B
    bool act = t < lens[b];
#pragma unroll
    for (int ct = 0; ct < 6; ++ct) {
      int nn = ct * 16 + n;
      if (nn < CC) {
        float val = act ? acc[ct][v] : 0.0f;
        float2 pr = make_float2(val, 1.0f - val);
        *(float2*)(out + (((size_t)b * TT + t) * CC + nn) * 2) = pr;
      }
    }
  }
}

// ---------------------------------------------------------------------------
extern "C" void kernel_launch(void* const* d_in, const int* in_sizes, int n_in,
                              void* d_out, int out_size, void* d_ws, size_t ws_size,
                              hipStream_t stream) {
  const float* x    = (const float*)d_in[0];   // input_sequences [T][B][F]
  const int*   lens = (const int*)  d_in[1];   // input_sequence_lengths [B]
  const float* Wih  = (const float*)d_in[2];   // [4H][F]
  const float* Whh  = (const float*)d_in[3];   // [4H][H]
  const float* bih  = (const float*)d_in[4];   // [4H]
  const float* bhh  = (const float*)d_in[5];   // [4H]
  const float* Wfc  = (const float*)d_in[6];   // [C][H]
  const float* bfc  = (const float*)d_in[7];   // [C]
  float* out = (float*)d_out;

  // Workspace layout (all 256B aligned):
  //   [ctr 256B][hbuf (T+1)*B*H bf16][wpack 64*4480 frags][fpack 6144 frags]
  unsigned* ctr  = (unsigned*)d_ws;
  __bf16*   hbuf = (__bf16*)((char*)d_ws + 256);
  size_t hbytes  = (size_t)(TT + 1) * BB * HH * sizeof(__bf16);
  v16bf* wpack   = (v16bf*)((char*)d_ws + 256 + hbytes);
  v16bf* fpack   = wpack + (size_t)NWG * REC_FRAGS;

  // 1) zero h_0 + barrier counter; pack weights into WMMA fragment blocks
  lstm_prep<<<(BB * HH + 255) / 256, 256, 0, stream>>>(hbuf, ctr);
  pack_rec_weights<<<(NWG * REC_FRAGS + 255) / 256, 256, 0, stream>>>(Whh, Wih,
                                                                      wpack);
  pack_fc_weights<<<(FC_FRAGS + 255) / 256, 256, 0, stream>>>(Wfc, fpack);

  // 2) persistent recurrence: W slices TDM-resident in LDS for all 1024 steps
  size_t lds_rec = (size_t)REC_FRAGS * sizeof(v16bf);   // 140 KB
  lstm_persistent<<<NWG, 256, lds_rec, stream>>>(x, lens, bih, bhh, wpack,
                                                 hbuf, ctr);

  // 3) FC head over the full h history (parallel WMMA GEMM)
  size_t lds_fc = (size_t)FC_FRAGS * sizeof(v16bf);     // 192 KB
  lstm_logits<<<(TT * BB / 16) / 8, 256, lds_fc, stream>>>(hbuf, lens, fpack,
                                                           bfc, out);
}
